// DirectedHypergraphNet_80290118631452
// MI455X (gfx1250) — matrix operand order repaired and checked by reading, not compiled, not run on hardware
//
// DirectedHypergraphNet for MI455X (gfx1250) — WMMA-based implementation, v2.
//
// All dense layers run through v_wmma_f32_16x16x32_f16 (f16 in, f32 acc).
// v2 GEMM changes vs v1 (driven by disasm):
//   - gather row pointer hoisted out of the K loop (was a dependent
//     global_load + wait every iteration),
//   - double-buffered LDS with software pipelining: global loads for tile
//     t+1 are issued before the WMMA block of tile t, LDS stores after, so
//     s_wait_loadcnt is covered by matrix math; one barrier per K-step,
//   - vectorized float4 staging + packed f32->f16 on interior tiles
//     (all hot GEMMs have K % 32 == 0), guarded scalar path only on tails,
//   - prefetch locality 3 (near) instead of SYS scope.
//
// Input order assumption: setup_inputs() dict insertion order, with the
// `params` pytree flattened in JAX order (dict keys alphabetical):
//   0..76 params leaves, 77 r_atom, 78 r_eidx, 79 r_efeat, 80 r_bidx,
//   81 p_atom, 82 p_eidx, 83 p_efeat, 84 p_bidx.
//
// Workspace requirement: ~268 MiB (checked; early-return if smaller).

#include <hip/hip_runtime.h>

typedef __attribute__((ext_vector_type(16))) _Float16 v16h;
typedef __attribute__((ext_vector_type(8)))  _Float16 v8h;
typedef __attribute__((ext_vector_type(8)))  float    v8f;

namespace {
constexpr int B_    = 512;    // NUM_GRAPHS
constexpr int HID   = 512;
constexpr int EDGED = 64;
constexpr int MOL   = 512;
constexpr int HDIM  = 768;
constexpr int NL    = 4;
constexpr int SLOTS = 3;
constexpr int NN    = 16384;
constexpr int NE    = 32768;
constexpr int ATOMF = 40;
constexpr int EDGEF = 12;
constexpr int NRX   = 10;
}

// ---------------------------------------------------------------------------
// Generic WMMA GEMM:  C = act( beta*C + gather(A) @ W + bias + bias2 )
//   A: [M x K] f32 (row stride lda); optional per-row gather via gidx.
//   W: [K x N] f32 row-major.  C: [M x N] f32.
//   act: 0 = identity, 1 = exact GELU.  beta: 0 or 1.
// Block: 256 threads = 8 waves; block tile 128x64; wave tile 32x32.
// Double-buffered LDS, software-pipelined global->LDS staging.
// ---------------------------------------------------------------------------
__global__ __launch_bounds__(256) void k_gemm(
    const float* __restrict__ A, const float* __restrict__ W,
    const float* __restrict__ bias, const float* __restrict__ bias2,
    float* __restrict__ C, const int* __restrict__ gidx,
    int M, int N, int K, int lda, int act, int beta)
{
  __shared__ __align__(16) _Float16 As[2][128][32];
  __shared__ __align__(16) _Float16 Bs[2][64][32];

  const int tid  = threadIdx.x;
  const int lane = tid & 31;
  const int wave = tid >> 5;       // 0..7
  const int wm   = wave >> 1;      // 0..3  (M direction)
  const int wn   = wave & 1;       // 0..1  (N direction)
  const int m0   = blockIdx.y * 128;
  const int n0   = blockIdx.x * 64;

  v8f acc[2][2] = {};

  // staging roles
  const int arow = tid >> 1;          // 0..127 : A tile row this thread stages
  const int akb  = (tid & 1) * 16;    // 0|16   : A k-chunk
  const int bn   = tid & 63;          // 0..63  : B tile column (n)
  const int bkb  = (tid >> 6) * 8;    // 0,8,16,24 : B k-chunk

  // fragment roles (CDNA5 16-bit WMMA lane layout)
  const int mrow = lane & 15;
  const int hi   = lane >> 4;         // lane half

  // ---- hoisted, loop-invariant pointers ----
  const float* arowp = nullptr;       // A row this thread stages (gathered)
  {
    int gm = m0 + arow;
    if (gm < M) {
      int src = gidx ? gidx[gm] : gm;
      arowp = A + (size_t)src * lda;
    }
  }
  const int   nn   = n0 + bn;
  const float* wcol = (nn < N) ? (W + nn) : nullptr;

  float ar[16];   // staged A values (f32), converted to f16 at LDS store
  float br[8];    // staged B values

  auto gload = [&](int k0) {
    if (arowp && (k0 + 32 <= K)) {            // interior: vectorized
      const float4* p = (const float4*)(arowp + k0 + akb);
      float4 f;
      f = p[0]; ar[0]  = f.x; ar[1]  = f.y; ar[2]  = f.z; ar[3]  = f.w;
      f = p[1]; ar[4]  = f.x; ar[5]  = f.y; ar[6]  = f.z; ar[7]  = f.w;
      f = p[2]; ar[8]  = f.x; ar[9]  = f.y; ar[10] = f.z; ar[11] = f.w;
      f = p[3]; ar[12] = f.x; ar[13] = f.y; ar[14] = f.z; ar[15] = f.w;
    } else {
#pragma unroll
      for (int j = 0; j < 16; ++j) {
        int kk = k0 + akb + j;
        ar[j] = (arowp && kk < K) ? arowp[kk] : 0.0f;
      }
    }
    if (wcol && (k0 + 32 <= K)) {
#pragma unroll
      for (int j = 0; j < 8; ++j) br[j] = wcol[(size_t)(k0 + bkb + j) * N];
    } else {
#pragma unroll
      for (int j = 0; j < 8; ++j) {
        int kk = k0 + bkb + j;
        br[j] = (wcol && kk < K) ? wcol[(size_t)kk * N] : 0.0f;
      }
    }
  };

  auto lstore = [&](int buf) {
    v8h h0, h1, hb;
#pragma unroll
    for (int j = 0; j < 8; ++j) {
      h0[j] = (_Float16)ar[j];
      h1[j] = (_Float16)ar[8 + j];
      hb[j] = (_Float16)br[j];
    }
    *(v8h*)&As[buf][arow][akb]     = h0;
    *(v8h*)&As[buf][arow][akb + 8] = h1;
    *(v8h*)&Bs[buf][bn][bkb]       = hb;
  };

  // ---- prologue: stage tile 0 ----
  gload(0);
  lstore(0);

  const int nk = (K + 31) >> 5;
  for (int t = 0; t < nk; ++t) {
    __syncthreads();                    // staged tile t visible; buf t^1 free
    const int cur = t & 1;

    // issue global loads for tile t+1 BEFORE the math (latency hiding)
    if (t + 1 < nk) {
      if (arowp && (t + 2) * 32 < K)
        __builtin_prefetch(arowp + (t + 2) * 32 + akb, 0, 3);
      gload((t + 1) * 32);
    }

    // ---- fragments per CDNA5 16-bit WMMA lane layout ----
    v16h afrag[2], bfrag[2];
#pragma unroll
    for (int mt = 0; mt < 2; ++mt) {
      const v8h* ap = (const v8h*)&As[cur][wm * 32 + mt * 16 + mrow][0];
      v8h c0 = ap[hi];        // lanes0-15: K 0..7   | lanes16-31: K 8..15
      v8h c1 = ap[2 + hi];    // lanes0-15: K 16..23 | lanes16-31: K 24..31
#pragma unroll
      for (int j = 0; j < 8; ++j) { afrag[mt][j] = c0[j]; afrag[mt][8 + j] = c1[j]; }
    }
#pragma unroll
    for (int nt = 0; nt < 2; ++nt) {
      const v8h* bp = (const v8h*)&Bs[cur][wn * 32 + nt * 16 + (lane & 15)][0];
      v8h c0 = bp[hi];        // lanes0-15: K 0..7   | lanes16-31: K 8..15
      v8h c1 = bp[2 + hi];    // lanes0-15: K 16..23 | lanes16-31: K 24..31
#pragma unroll
      for (int j = 0; j < 8; ++j) { bfrag[nt][j] = c0[j]; bfrag[nt][8 + j] = c1[j]; }
    }

#pragma unroll
    for (int mt = 0; mt < 2; ++mt)
#pragma unroll
      for (int nt = 0; nt < 2; ++nt)
        acc[mt][nt] = __builtin_amdgcn_wmma_f32_16x16x32_f16(
            false, afrag[mt], false, bfrag[nt],
            (short)0, acc[mt][nt], false, false);

    // store staged tile t+1 into the other buffer (wait_loadcnt lands here,
    // after the WMMA block has covered the global latency)
    if (t + 1 < nk) lstore(cur ^ 1);
  }

  // ---- epilogue: C/D layout — vgpr v, lane l -> row = v + 8*hi, col = l&15 ----
#pragma unroll
  for (int mt = 0; mt < 2; ++mt)
#pragma unroll
    for (int nt = 0; nt < 2; ++nt)
#pragma unroll
      for (int v = 0; v < 8; ++v) {
        int row = m0 + wm * 32 + mt * 16 + v + hi * 8;
        int col = n0 + wn * 32 + nt * 16 + (lane & 15);
        if (row < M && col < N) {
          float val = acc[mt][nt][v];
          if (bias)  val += bias[col];
          if (bias2) val += bias2[col];
          size_t o = (size_t)row * N + col;
          if (beta)  val += C[o];
          if (act == 1) val = 0.5f * val * (1.0f + erff(val * 0.70710678118654752f));
          C[o] = val;
        }
      }
}

// ---------------------------------------------------------------------------
__global__ void k_zero(float* __restrict__ p, long long n) {
  long long i  = (long long)blockIdx.x * blockDim.x + threadIdx.x;
  long long st = (long long)gridDim.x * blockDim.x;
  for (; i < n; i += st) p[i] = 0.0f;
}

__global__ void k_scatter_add(const float* __restrict__ msg, const int* __restrict__ idx,
                              float* __restrict__ agg, int E, int D) {
  int per = D >> 2;
  int t = blockIdx.x * blockDim.x + threadIdx.x;
  if (t >= E * per) return;
  int e = t / per, c = (t - e * per) * 4;
  int n = idx[e];
  const float* m = msg + (size_t)e * D + c;
  float*       a = agg + (size_t)n * D + c;
  atomicAdd(a + 0, m[0]); atomicAdd(a + 1, m[1]);
  atomicAdd(a + 2, m[2]); atomicAdd(a + 3, m[3]);
}

__global__ void k_count(const int* __restrict__ idx, float* __restrict__ cnt, int E) {
  int e = blockIdx.x * blockDim.x + threadIdx.x;
  if (e < E) atomicAdd(&cnt[idx[e]], 1.0f);
}

__global__ void k_div_rows(float* __restrict__ x, const float* __restrict__ cnt,
                           int R, int D) {
  int t = blockIdx.x * blockDim.x + threadIdx.x;
  if (t >= R * D) return;
  float c = cnt[t / D];
  x[t] /= (c < 1.0f ? 1.0f : c);
}

// y = LayerNorm(x + r) * g + b   (one wave32 per row; safe for y == x)
__global__ __launch_bounds__(256) void k_ln(
    const float* __restrict__ x, const float* __restrict__ r,
    const float* __restrict__ g, const float* __restrict__ b,
    float* __restrict__ y, int M, int D)
{
  int lane = threadIdx.x & 31;
  int row  = blockIdx.x * 8 + (threadIdx.x >> 5);
  if (row >= M) return;
  const float* xr = x + (size_t)row * D;
  const float* rr = r ? r + (size_t)row * D : nullptr;
  float s = 0.0f, sq = 0.0f;
  for (int i = lane; i < D; i += 32) {
    float v = xr[i] + (rr ? rr[i] : 0.0f);
    s += v; sq += v * v;
  }
#pragma unroll
  for (int o = 16; o > 0; o >>= 1) {
    s  += __shfl_xor(s,  o, 32);
    sq += __shfl_xor(sq, o, 32);
  }
  float mean = s / D;
  float var  = sq / D - mean * mean;
  float inv  = rsqrtf(var + 1e-5f);
  float* yr  = y + (size_t)row * D;
  for (int i = lane; i < D; i += 32) {
    float v = xr[i] + (rr ? rr[i] : 0.0f);
    yr[i] = (v - mean) * inv * g[i] + b[i];
  }
}

// 12-head attention over S=3 tokens; qkv [B,3,2304] -> o [B,3,768]
__global__ void k_attn(const float* __restrict__ qkv, float* __restrict__ o, int B) {
  int t = blockIdx.x * blockDim.x + threadIdx.x;
  if (t >= B * 12) return;
  int b = t / 12, h = t - b * 12;
  const float* base = qkv + (size_t)b * 3 * 2304;
  float sc[3][3];
#pragma unroll
  for (int i = 0; i < 3; ++i)
#pragma unroll
    for (int j = 0; j < 3; ++j) {
      const float* q = base + i * 2304 + h * 64;
      const float* k = base + j * 2304 + 768 + h * 64;
      float a = 0.0f;
      for (int d = 0; d < 64; ++d) a += q[d] * k[d];
      sc[i][j] = a * 0.125f;   // 1/sqrt(64)
    }
#pragma unroll
  for (int i = 0; i < 3; ++i) {
    float mx = fmaxf(sc[i][0], fmaxf(sc[i][1], sc[i][2]));
    float e0 = expf(sc[i][0] - mx), e1 = expf(sc[i][1] - mx), e2 = expf(sc[i][2] - mx);
    float inv = 1.0f / (e0 + e1 + e2);
    sc[i][0] = e0 * inv; sc[i][1] = e1 * inv; sc[i][2] = e2 * inv;
  }
  const float* v0 = base + 0 * 2304 + 1536 + h * 64;
  const float* v1 = base + 1 * 2304 + 1536 + h * 64;
  const float* v2 = base + 2 * 2304 + 1536 + h * 64;
#pragma unroll
  for (int i = 0; i < 3; ++i) {
    float* op = o + ((size_t)b * 3 + i) * 768 + h * 64;
    for (int d = 0; d < 64; ++d)
      op[d] = sc[i][0] * v0[d] + sc[i][1] * v1[d] + sc[i][2] * v2[d];
  }
}

__global__ void k_mean3(const float* __restrict__ x, float* __restrict__ y, int B, int D) {
  int t = blockIdx.x * blockDim.x + threadIdx.x;
  if (t >= B * D) return;
  int b = t / D, d = t - b * D;
  const float* p = x + (size_t)b * 3 * D + d;
  y[t] = (p[0] + p[D] + p[2 * D]) * (1.0f / 3.0f);
}

// x [S*B, D] (slot-major) -> y [B*S, D] (batch-major)
__global__ void k_sb2bs(const float* __restrict__ x, float* __restrict__ y,
                        int S, int B, int D) {
  long long t = (long long)blockIdx.x * blockDim.x + threadIdx.x;
  if (t >= (long long)S * B * D) return;
  int d = (int)(t % D);
  long long rest = t / D;
  int b = (int)(rest % B), s = (int)(rest / B);
  y[((size_t)b * S + s) * D + d] = x[t];
}

// ---------------------------------------------------------------------------
// Host orchestration
// ---------------------------------------------------------------------------
namespace {

inline void gemm(hipStream_t s, const float* A, const float* W, const float* bias,
                 const float* bias2, float* C, const int* gidx,
                 int M, int N, int K, int lda, int act, int beta) {
  dim3 grid((N + 63) / 64, (M + 127) / 128);
  k_gemm<<<grid, 256, 0, s>>>(A, W, bias, bias2, C, gidx, M, N, K, lda, act, beta);
}

struct EncPtrs {
  const float *apb, *apw, *epb, *epw, *opb, *opw;
  const float *lnb[NL], *lng[NL], *m1b[NL], *m1w[NL], *m2b[NL], *m2w[NL],
              *u1b[NL], *u1w[NL], *u2b[NL], *u2w[NL];
};

// Workspace layout (offsets in floats)
constexpr size_t OX    = 0;
constexpr size_t OEP   = OX    + (size_t)NN * HID;         //  8,388,608
constexpr size_t OBIG  = OEP   + (size_t)NE * EDGED;       // +2,097,152
constexpr size_t OMSG  = OBIG  + (size_t)NE * 2 * HID;     // +33,554,432
constexpr size_t OAGG  = OMSG  + (size_t)NE * HID;         // +16,777,216
constexpr size_t OCNT  = OAGG  + (size_t)NN * HID;         // +8,388,608
constexpr size_t OPCNT = OCNT  + NN;
constexpr size_t OPOOL = OPCNT + 1024;
constexpr size_t OREMB = OPOOL + (size_t)B_ * HID;
constexpr size_t OEND  = OREMB + (size_t)SLOTS * B_ * MOL; // ~70.3M floats

void run_encoder(hipStream_t s, const EncPtrs& P,
                 const float* atom, const int* erow, const int* ecol,
                 const float* efeat, const int* bidx, float* ws, float* emb)
{
  float* X    = ws + OX;
  float* EP   = ws + OEP;
  float* BIG  = ws + OBIG;
  float* MSG  = ws + OMSG;
  float* AGG  = ws + OAGG;
  float* CNT  = ws + OCNT;
  float* PCNT = ws + OPCNT;
  float* POOL = ws + OPOOL;
  float* H1 = BIG;                       // [NE, 2*HID]
  float* UH = BIG;                       // reuses H1 after MSG is built
  float* UO = BIG + (size_t)NN * HID;

  gemm(s, atom,  P.apw, P.apb, nullptr, X,  nullptr, NN, HID,   ATOMF, ATOMF, 0, 0);
  gemm(s, efeat, P.epw, P.epb, nullptr, EP, nullptr, NE, EDGED, EDGEF, EDGEF, 0, 0);

  for (int l = 0; l < NL; ++l) {
    // msg = gelu([x[row], x[col], e] @ W1 + b1) @ W2 + b2  — concat as 3 passes
    gemm(s, X,  P.m1w[l],                           P.m1b[l], nullptr, H1, erow,    NE, 2*HID, HID,   HID,   0, 0);
    gemm(s, X,  P.m1w[l] + (size_t)HID * 2 * HID,   nullptr,  nullptr, H1, ecol,    NE, 2*HID, HID,   HID,   0, 1);
    gemm(s, EP, P.m1w[l] + (size_t)2*HID * 2 * HID, nullptr,  nullptr, H1, nullptr, NE, 2*HID, EDGED, EDGED, 1, 1);
    gemm(s, H1, P.m2w[l], P.m2b[l], nullptr, MSG, nullptr, NE, HID, 2*HID, 2*HID, 0, 0);

    // aggr = segment_mean(msg, col, NN)
    k_zero<<<2048, 256, 0, s>>>(AGG, (long long)NN * HID);
    k_zero<<<64,   256, 0, s>>>(CNT, NN);
    k_scatter_add<<<(NE * (HID / 4) + 255) / 256, 256, 0, s>>>(MSG, ecol, AGG, NE, HID);
    k_count<<<(NE + 255) / 256, 256, 0, s>>>(ecol, CNT, NE);
    k_div_rows<<<(NN * HID + 255) / 256, 256, 0, s>>>(AGG, CNT, NN, HID);

    // x = LN(x + upd2(gelu(upd1([x, aggr]))))
    gemm(s, X,   P.u1w[l],                      P.u1b[l], nullptr, UH, nullptr, NN, HID, HID, HID, 0, 0);
    gemm(s, AGG, P.u1w[l] + (size_t)HID * HID,  nullptr,  nullptr, UH, nullptr, NN, HID, HID, HID, 1, 1);
    gemm(s, UH,  P.u2w[l], P.u2b[l], nullptr, UO, nullptr, NN, HID, HID, HID, 0, 0);
    k_ln<<<(NN + 7) / 8, 256, 0, s>>>(X, UO, P.lng[l], P.lnb[l], X, NN, HID);
  }

  // pooled = segment_mean(x, batch_idx, B); emb = pooled @ out_proj
  k_zero<<<1024, 256, 0, s>>>(POOL, (long long)B_ * HID);
  k_zero<<<4,    256, 0, s>>>(PCNT, B_);
  k_scatter_add<<<(NN * (HID / 4) + 255) / 256, 256, 0, s>>>(X, bidx, POOL, NN, HID);
  k_count<<<(NN + 255) / 256, 256, 0, s>>>(bidx, PCNT, NN);
  k_div_rows<<<(B_ * HID + 255) / 256, 256, 0, s>>>(POOL, PCNT, B_, HID);
  gemm(s, POOL, P.opw, P.opb, nullptr, emb, nullptr, B_, MOL, HID, HID, 0, 0);
}

} // namespace

extern "C" void kernel_launch(void* const* d_in, const int* in_sizes, int n_in,
                              void* d_out, int out_size, void* d_ws, size_t ws_size,
                              hipStream_t stream)
{
  (void)in_sizes; (void)n_in; (void)out_size;
  if (ws_size < OEND * sizeof(float)) return;  // needs ~268 MiB scratch

  auto f  = [&](int i) { return (const float*)d_in[i]; };
  auto iv = [&](int i) { return (const int*)d_in[i]; };

  // ---- params leaves (JAX pytree flatten order; dict keys alphabetical) ----
  const float *co0b = f(0),  *co0w = f(1),  *co1b = f(2),  *co1w = f(3);
  EncPtrs P;
  P.apb = f(4); P.apw = f(5); P.epb = f(6); P.epw = f(7);
  for (int l = 0; l < NL; ++l) {
    int b0 = 8 + l * 10;   // per-layer: ln.b, ln.g, msg1.b, msg1.w, msg2.b,
                           //            msg2.w, upd1.b, upd1.w, upd2.b, upd2.w
    P.lnb[l] = f(b0 + 0); P.lng[l] = f(b0 + 1);
    P.m1b[l] = f(b0 + 2); P.m1w[l] = f(b0 + 3);
    P.m2b[l] = f(b0 + 4); P.m2w[l] = f(b0 + 5);
    P.u1b[l] = f(b0 + 6); P.u1w[l] = f(b0 + 7);
    P.u2b[l] = f(b0 + 8); P.u2w[l] = f(b0 + 9);
  }
  P.opb = f(48); P.opw = f(49);
  const float *ipb  = f(50), *ipw  = f(51);
  const float *pr0b = f(52), *pr0w = f(53), *pr1b = f(54), *pr1w = f(55);
  const float *role = f(56);                                // role_emb[0] = first 768
  const float *rx0b = f(57), *rx0w = f(58), *rx1b = f(59), *rx1w = f(60);
  const float *sc0b = f(61), *sc0w = f(62), *sc1b = f(63), *sc1w = f(64);
  const float *ff1b = f(65), *ff1w = f(66), *ff2b = f(67), *ff2w = f(68);
  const float *qkvb = f(69), *qkvw = f(70);
  const float *ln1b = f(71), *ln1g = f(72), *ln2b = f(73), *ln2g = f(74);
  const float *topb = f(75), *topw = f(76);

  const float* r_atom  = f(77);  const int* r_eidx = iv(78);
  const float* r_efeat = f(79);  const int* r_bidx = iv(80);
  const float* p_atom  = f(81);  const int* p_eidx = iv(82);
  const float* p_efeat = f(83);  const int* p_bidx = iv(84);

  float* ws   = (float*)d_ws;
  float* REMB = ws + OREMB;
  float* BIG  = ws + OBIG;

  float* out    = (float*)d_out;
  float* out_co = out;                                  // [B,S,MOL]
  float* out_pp = out + (size_t)B_ * SLOTS * MOL;       // [B,MOL]  @ 786432
  float* out_pe = out_pp + (size_t)B_ * MOL;            // [B,MOL]  @ 1048576
  float* out_rx = out_pe + (size_t)B_ * MOL;            // [B,NRX]  @ 1310720
  float* out_sc = out_rx + (size_t)B_ * NRX;            // [B,1]    @ 1315840

  // ---- Stage 1: product encoder first (frees BIG for stage 2), then slots ----
  run_encoder(stream, P, p_atom, p_eidx, p_eidx + NE, p_efeat, p_bidx, ws, out_pe);
  for (int sl = 0; sl < SLOTS; ++sl)
    run_encoder(stream, P,
                r_atom  + (size_t)sl * NN * ATOMF,
                r_eidx  + (size_t)sl * 2 * NE,
                r_eidx  + (size_t)sl * 2 * NE + NE,
                r_efeat + (size_t)sl * NE * EDGEF,
                r_bidx  + (size_t)sl * NN,
                ws, REMB + (size_t)sl * B_ * MOL);

  // ---- Stage 2 buffers packed inside BIG (encoder scratch now dead) ----
  float* HTMP  = BIG;                      // [S*B, HDIM]
  float* HST   = HTMP  + (size_t)1179648;  // [B*S, HDIM]
  float* QKV   = HST   + (size_t)1179648;  // [B*S, 3*HDIM]
  float* ATTO  = QKV   + (size_t)3538944;  // [B*S, HDIM]
  float* AO    = ATTO  + (size_t)1179648;
  float* X1    = AO    + (size_t)1179648;
  float* F1    = X1    + (size_t)1179648;  // [B*S, 2*HDIM]
  float* F2    = F1    + (size_t)2359296;
  float* X2    = F2    + (size_t)1179648;
  float* TAILR = X2    + (size_t)1179648;  // [B, HDIM]
  float* HH    = TAILR + (size_t)393216;   // head hidden scratch

  const int BS = SLOTS * B_;  // 1536

  // h = input_proj(r_embs) + role_emb[0]; transpose slot-major -> batch-major
  gemm(stream, REMB, ipw, ipb, role, HTMP, nullptr, BS, HDIM, MOL, MOL, 0, 0);
  k_sb2bs<<<((long long)BS * HDIM + 255) / 256, 256, 0, stream>>>(HTMP, HST, SLOTS, B_, HDIM);

  // co head (on h_stack, before tail encoder)
  gemm(stream, HST, co0w, co0b, nullptr, HH,     nullptr, BS, HDIM, HDIM, HDIM, 1, 0);
  gemm(stream, HH,  co1w, co1b, nullptr, out_co, nullptr, BS, MOL,  HDIM, HDIM, 0, 0);

  // TailSetEncoder: MHA + FFN with post-residual LayerNorms
  gemm(stream, HST, qkvw, qkvb, nullptr, QKV, nullptr, BS, 3 * HDIM, HDIM, HDIM, 0, 0);
  k_attn<<<(B_ * 12 + 255) / 256, 256, 0, stream>>>(QKV, ATTO, B_);
  gemm(stream, ATTO, topw, topb, nullptr, AO, nullptr, BS, HDIM, HDIM, HDIM, 0, 0);
  k_ln<<<(BS + 7) / 8, 256, 0, stream>>>(HST, AO, ln1g, ln1b, X1, BS, HDIM);
  gemm(stream, X1, ff1w, ff1b, nullptr, F1, nullptr, BS, 2 * HDIM, HDIM,     HDIM,     1, 0);
  gemm(stream, F1, ff2w, ff2b, nullptr, F2, nullptr, BS, HDIM,     2 * HDIM, 2 * HDIM, 0, 0);
  k_ln<<<(BS + 7) / 8, 256, 0, stream>>>(X1, F2, ln2g, ln2b, X2, BS, HDIM);
  k_mean3<<<(B_ * HDIM + 255) / 256, 256, 0, stream>>>(X2, TAILR, B_, HDIM);

  // heads off tail_repr
  gemm(stream, TAILR, pr0w, pr0b, nullptr, HH,     nullptr, B_, HDIM,     HDIM,     HDIM,     1, 0);
  gemm(stream, HH,    pr1w, pr1b, nullptr, out_pp, nullptr, B_, MOL,      HDIM,     HDIM,     0, 0);
  gemm(stream, TAILR, rx0w, rx0b, nullptr, HH,     nullptr, B_, HDIM / 2, HDIM,     HDIM,     1, 0);
  gemm(stream, HH,    rx1w, rx1b, nullptr, out_rx, nullptr, B_, NRX,      HDIM / 2, HDIM / 2, 0, 0);
  gemm(stream, TAILR, sc0w, sc0b, nullptr, HH,     nullptr, B_, HDIM / 2, HDIM,     HDIM,     1, 0);
  gemm(stream, HH,    sc1w, sc1b, nullptr, out_sc, nullptr, B_, 1,        HDIM / 2, HDIM / 2, 0, 0);
}
